// ClusterMemory_28535762714738
// MI455X (gfx1250) — compile-verified, matrix operand order checked by reading, not run on hardware
//
#include <hip/hip_runtime.h>
#include <math.h>

// Problem constants (match reference)
#define NS    65536        // num_samples (N of GEMM)
#define NF    256          // num_features (K of GEMM)
#define BATCH 512          // batch (M of GEMM)
#define TEMP_INV 20.0f     // 1 / 0.05
#define MOM   0.2f
#define EPSN  1e-12f

typedef __attribute__((ext_vector_type(16))) __bf16 v16bf;
typedef __attribute__((ext_vector_type(8)))  __bf16 v8bf;
typedef __attribute__((ext_vector_type(8)))  float  v8f;
typedef __attribute__((ext_vector_type(4)))  float  v4f;

__device__ __forceinline__ v16bf cat8(v8bf a, v8bf b) {
  return __builtin_shufflevector(a, b, 0,1,2,3,4,5,6,7,8,9,10,11,12,13,14,15);
}

// Split fp32 -> bf16 hi + bf16 lo (x ~= hi + lo, ~16 mantissa bits retained)
__device__ __forceinline__ void split16(const float* __restrict__ f, v16bf& hi, v16bf& lo) {
  #pragma unroll
  for (int i = 0; i < 16; ++i) {
    float x = f[i];
    __bf16 h = (__bf16)x;
    hi[i] = h;
    lo[i] = (__bf16)(x - (float)h);
  }
}

// ---------------------------------------------------------------------------
// Kernel 1: logits = (inputs @ features^T) * 20 via bf16 split WMMA.
// Block = 256 threads = 8 waves; block owns a 128-column slab of N.
// Each wave holds its 16-col B tile (bf16 hi/lo) in registers across all
// 32 M-tiles; A tile staged/converted once per block in LDS.
// ---------------------------------------------------------------------------
__global__ __launch_bounds__(256) void cm_logits_wmma(
    const float* __restrict__ inp,   // [512][256]
    const float* __restrict__ feat,  // [65536][256]
    float* __restrict__ out)         // [512][65536]
{
  const int tid   = threadIdx.x;
  const int wave  = tid >> 5;
  const int lane  = tid & 31;
  const int laneN = lane & 15;      // row (A) / col (B) index within tile
  const int laneH = lane >> 4;      // half-wave selects K sub-offset (+8)
  const int n0    = blockIdx.x * 128 + wave * 16;

  // --- Load & split this wave's B tile: 16 feature rows x 256 K ----------
  // ISA 16-bit operand layout: lane<16 holds K={c*32..+7, c*32+16..+23},
  // lane>=16 holds the +8 shifted chunks.
  v16bf Bhi[8], Blo[8];
  {
    const float* brow = feat + (size_t)(n0 + laneN) * NF;
    #pragma unroll
    for (int c = 0; c < 8; ++c) {
      const int k0 = c * 32 + laneH * 8;
      float f[16] __attribute__((aligned(16)));
      *(v4f*)(f + 0)  = *(const v4f*)(brow + k0);
      *(v4f*)(f + 4)  = *(const v4f*)(brow + k0 + 4);
      *(v4f*)(f + 8)  = *(const v4f*)(brow + k0 + 16);
      *(v4f*)(f + 12) = *(const v4f*)(brow + k0 + 20);
      split16(f, Bhi[c], Blo[c]);
    }
  }

  // A tile staged as bf16 hi/lo in LDS; +8 element row pad keeps 16B
  // alignment and spreads rows across banks (528B stride -> bank step 4).
  __shared__ __align__(16) __bf16 sAhi[16][264];
  __shared__ __align__(16) __bf16 sAlo[16][264];

  for (int mt = 0; mt < 32; ++mt) {
    __syncthreads();
    // Cooperative fp32 -> bf16 hi/lo conversion of the 16x256 A tile.
    {
      const int r    = tid >> 4;
      const int cseg = (tid & 15) * 16;
      const float* arow = inp + (size_t)(mt * 16 + r) * NF + cseg;
      float f[16] __attribute__((aligned(16)));
      *(v4f*)(f + 0)  = *(const v4f*)(arow + 0);
      *(v4f*)(f + 4)  = *(const v4f*)(arow + 4);
      *(v4f*)(f + 8)  = *(const v4f*)(arow + 8);
      *(v4f*)(f + 12) = *(const v4f*)(arow + 12);
      v8bf h0, h1, l0, l1;
      #pragma unroll
      for (int i = 0; i < 8; ++i) {
        __bf16 ha = (__bf16)f[i];
        __bf16 hb = (__bf16)f[8 + i];
        h0[i] = ha; l0[i] = (__bf16)(f[i]     - (float)ha);
        h1[i] = hb; l1[i] = (__bf16)(f[8 + i] - (float)hb);
      }
      *(v8bf*)&sAhi[r][cseg]     = h0;
      *(v8bf*)&sAhi[r][cseg + 8] = h1;
      *(v8bf*)&sAlo[r][cseg]     = l0;
      *(v8bf*)&sAlo[r][cseg + 8] = l1;
    }
    __syncthreads();

    v8f acc = {0.f, 0.f, 0.f, 0.f, 0.f, 0.f, 0.f, 0.f};
    #pragma unroll
    for (int c = 0; c < 8; ++c) {
      const int k0 = c * 32 + laneH * 8;
      v16bf ahi = cat8(*(const v8bf*)&sAhi[laneN][k0],
                       *(const v8bf*)&sAhi[laneN][k0 + 16]);
      v16bf alo = cat8(*(const v8bf*)&sAlo[laneN][k0],
                       *(const v8bf*)&sAlo[laneN][k0 + 16]);
      // hi*hi + hi*lo + lo*hi : ~fp32-accurate product, all bf16 WMMA rate
      acc = __builtin_amdgcn_wmma_f32_16x16x32_bf16(false, ahi, false, Bhi[c],
                                                    (short)0, acc, false, false);
      acc = __builtin_amdgcn_wmma_f32_16x16x32_bf16(false, ahi, false, Blo[c],
                                                    (short)0, acc, false, false);
      acc = __builtin_amdgcn_wmma_f32_16x16x32_bf16(false, alo, false, Bhi[c],
                                                    (short)0, acc, false, false);
    }

    // C/D layout: VGPR v, lane l -> (M = v + (l>>4)*8, N = l&15)
    float* orow = out + (size_t)(mt * 16 + laneH * 8) * NS + n0 + laneN;
    #pragma unroll
    for (int v = 0; v < 8; ++v)
      orow[(size_t)v * NS] = acc[v] * TEMP_INV;
  }
}

// ---------------------------------------------------------------------------
// Kernel 2: bulk copy features -> new_features (rows later overwritten by
// the scatter kernel, which runs after on the same stream).
// ---------------------------------------------------------------------------
__global__ __launch_bounds__(256) void cm_copy(const float* __restrict__ src,
                                               float* __restrict__ dst, int n4) {
  int i = blockIdx.x * blockDim.x + threadIdx.x;
  const int stride = gridDim.x * blockDim.x;
  const v4f* s = (const v4f*)src;
  v4f* d = (v4f*)dst;
  for (; i < n4; i += stride) d[i] = s[i];
}

// ---------------------------------------------------------------------------
// Kernel 3: sequential EMA scatter update. One wave per batch index; only
// the first occurrence of a target walks its (rare) duplicate chain in
// batch order, matching lax.scan semantics exactly.
// ---------------------------------------------------------------------------
__global__ __launch_bounds__(256) void cm_update(
    const float* __restrict__ inp, const int* __restrict__ tgt,
    const float* __restrict__ feat, float* __restrict__ outF)
{
  const int w    = blockIdx.x * 8 + (threadIdx.x >> 5);
  const int lane = threadIdx.x & 31;
  const int y    = tgt[w];

  // Bail if an earlier batch element targets the same row (chain owner wins).
  for (int j = 0; j < w; ++j)
    if (tgt[j] == y) return;

  float r[8];
  #pragma unroll
  for (int c = 0; c < 8; ++c)
    r[c] = feat[(size_t)y * NF + lane + c * 32];

  for (int j = w; j < BATCH; ++j) {
    if (tgt[j] != y) continue;
    const float* x = inp + (size_t)j * NF;
    float ss = 0.f;
    #pragma unroll
    for (int c = 0; c < 8; ++c) {
      r[c] = MOM * r[c] + (1.0f - MOM) * x[lane + c * 32];
      ss += r[c] * r[c];
    }
    #pragma unroll
    for (int off = 16; off > 0; off >>= 1)
      ss += __shfl_xor(ss, off, 32);
    const float inv = 1.0f / sqrtf(ss + EPSN);
    #pragma unroll
    for (int c = 0; c < 8; ++c) r[c] *= inv;
  }

  #pragma unroll
  for (int c = 0; c < 8; ++c)
    outF[(size_t)y * NF + lane + c * 32] = r[c];
}

extern "C" void kernel_launch(void* const* d_in, const int* in_sizes, int n_in,
                              void* d_out, int out_size, void* d_ws, size_t ws_size,
                              hipStream_t stream) {
  const float* inp  = (const float*)d_in[0];   // [512,256]
  const int*   tgt  = (const int*)d_in[1];     // [512]
  const float* feat = (const float*)d_in[2];   // [65536,256]
  float* logits = (float*)d_out;                       // [512,65536]
  float* outF   = logits + (size_t)BATCH * NS;         // [65536,256]

  cm_logits_wmma<<<dim3(NS / 128), 256, 0, stream>>>(inp, feat, logits);
  cm_copy<<<4096, 256, 0, stream>>>(feat, outF, NS * NF / 4);
  cm_update<<<BATCH / 8, 256, 0, stream>>>(inp, tgt, feat, outF);
}
